// MultiHeadAttention_3582002725550
// MI455X (gfx1250) — matrix operand order repaired and checked by reading, not compiled, not run on hardware
//
#include <hip/hip_runtime.h>
#include <hip/hip_bf16.h>

// ---- problem constants (B=2, S=2048, D=1024, H=16, DK=64) ----
#define S_LEN 2048
#define D_DIM 1024
#define H_NUM 16
#define DKH   64
#define BATCH 2
#define M_TOT (BATCH * S_LEN) // 4096

typedef __attribute__((ext_vector_type(16))) _Float16 v16h;
typedef __attribute__((ext_vector_type(8)))  _Float16 v8h;
typedef __attribute__((ext_vector_type(4)))  _Float16 v4h;
typedef __attribute__((ext_vector_type(8)))  float    v8f;

__device__ __forceinline__ v16h cat8(v8h lo, v8h hi) {
  return __builtin_shufflevector(lo, hi, 0,1,2,3,4,5,6,7,8,9,10,11,12,13,14,15);
}

__device__ __forceinline__ v8f wmma16(v16h a, v16h b, v8f c) {
  // D = A(16x32 f16) * B(32x16 f16) + C(16x16 f32)
  return __builtin_amdgcn_wmma_f32_16x16x32_f16(
      /*neg_a=*/false, a, /*neg_b=*/false, b,
      /*c_mod=*/(short)0, c, /*reuse_a=*/false, /*reuse_b=*/false);
}

__device__ __forceinline__ float rowmax16(float x) {
  x = fmaxf(x, __shfl_xor(x, 1, 32));
  x = fmaxf(x, __shfl_xor(x, 2, 32));
  x = fmaxf(x, __shfl_xor(x, 4, 32));
  x = fmaxf(x, __shfl_xor(x, 8, 32));
  return x;
}
__device__ __forceinline__ float rowsum16(float x) {
  x += __shfl_xor(x, 1, 32);
  x += __shfl_xor(x, 2, 32);
  x += __shfl_xor(x, 4, 32);
  x += __shfl_xor(x, 8, 32);
  return x;
}

// ---------------- kernel 1: cast q/k/v f32 -> f16, row-major [3][4096][1024]
__global__ __launch_bounds__(256) void cvt_x_kernel(
    const float* __restrict__ q, const float* __restrict__ k,
    const float* __restrict__ v, _Float16* __restrict__ x16) {
  size_t t = (size_t)blockIdx.x * 256 + threadIdx.x;
  size_t i = t * 4;
  const size_t per = (size_t)M_TOT * D_DIM; // 4194304
  const float* src; size_t off;
  if (i < per)            { src = q; off = i; }
  else if (i < 2 * per)   { src = k; off = i - per; }
  else                    { src = v; off = i - 2 * per; }
  float4 f = *(const float4*)(src + off);
  v4h hv;
  hv.x = (_Float16)f.x; hv.y = (_Float16)f.y;
  hv.z = (_Float16)f.z; hv.w = (_Float16)f.w;
  *(v4h*)(x16 + i) = hv;
}

// ---------------- kernel 2: cast+transpose W f32[K][N] -> f16 Wt[3][N][K]
__global__ __launch_bounds__(256) void cvt_w_kernel(
    const float* __restrict__ Wq, const float* __restrict__ Wk,
    const float* __restrict__ Wv, _Float16* __restrict__ w16t) {
  size_t i = (size_t)blockIdx.x * 256 + threadIdx.x; // [which][n][k]
  const size_t per = (size_t)D_DIM * D_DIM;
  const float* W = (i < per) ? Wq : ((i < 2 * per) ? Wk : Wv);
  size_t r = i % per;
  size_t n = r >> 10, kk = r & 1023;
  w16t[i] = (_Float16)W[kk * D_DIM + n];
}

// ---------------- kernel 3: QKV projection GEMM via WMMA
// Wave tile 32x64: two A fragments reuse four B fragments -> 8 WMMA per
// 192B of fragment loads (24 B/WMMA operand traffic). Writes Q/K as
// [B,H,S,DK] f16, V transposed as [B,H,DK,S] f16.
__global__ __launch_bounds__(256) void qkv_gemm_kernel(
    const _Float16* __restrict__ x16, const _Float16* __restrict__ w16t,
    const float* __restrict__ bq, const float* __restrict__ bk,
    const float* __restrict__ bv,
    _Float16* __restrict__ qh, _Float16* __restrict__ kh,
    _Float16* __restrict__ vhT) {
  int which = blockIdx.z;
  const _Float16* X  = x16  + (size_t)which * M_TOT * D_DIM;
  const _Float16* Wt = w16t + (size_t)which * D_DIM * D_DIM;
  const float* bias  = (which == 0) ? bq : ((which == 1) ? bk : bv);
  _Float16* dst      = (which == 0) ? qh : ((which == 1) ? kh : vhT);

  int lane = threadIdx.x & 31;
  int wave = threadIdx.x >> 5;
  int g = lane >> 4, ln = lane & 15;
  int m0 = blockIdx.x * 256 + wave * 32; // 32 rows per wave
  int n0 = blockIdx.y * 64;

  const _Float16* xrow0 = X + (size_t)(m0 + ln) * D_DIM;
  const _Float16* xrow1 = X + (size_t)(m0 + 16 + ln) * D_DIM;
  const _Float16* wr[4];
#pragma unroll
  for (int nt = 0; nt < 4; ++nt)
    wr[nt] = Wt + (size_t)(n0 + nt * 16 + ln) * D_DIM;

  v8f acc0[4] = {};
  v8f acc1[4] = {};
  for (int k0 = 0; k0 < D_DIM; k0 += 32) {
    // A fragments: row = ln (+16), K = g*16 + 0..15
    v16h af0 = cat8(*(const v8h*)(xrow0 + k0 + g * 16),
                    *(const v8h*)(xrow0 + k0 + g * 16 + 8));
    v16h af1 = cat8(*(const v8h*)(xrow1 + k0 + g * 16),
                    *(const v8h*)(xrow1 + k0 + g * 16 + 8));
#pragma unroll
    for (int nt = 0; nt < 4; ++nt) {
      // B fragment: col = ln, K = g*16 + 0..15 (contiguous in Wt row)
      v16h bf = *(const v16h*)(wr[nt] + k0 + g * 16);
      acc0[nt] = wmma16(af0, bf, acc0[nt]);
      acc1[nt] = wmma16(af1, bf, acc1[nt]);
    }
  }

#pragma unroll
  for (int nt = 0; nt < 4; ++nt) {
    int n = n0 + nt * 16 + ln;
    float bn = bias[n];
    int h = n >> 6, dk = n & 63;
#pragma unroll
    for (int mt = 0; mt < 2; ++mt) {
#pragma unroll
      for (int r = 0; r < 8; ++r) {
        int m = m0 + mt * 16 + r + 8 * g;
        int b = m >> 11, s = m & 2047;
        float val = (mt == 0 ? acc0[nt][r] : acc1[nt][r]) + bn;
        size_t idx;
        if (which == 2)
          idx = (((size_t)(b * H_NUM + h)) * DKH + dk) * S_LEN + s; // V^T
        else
          idx = (((size_t)(b * H_NUM + h)) * S_LEN + s) * DKH + dk; // Q, K
        dst[idx] = (_Float16)val;
      }
    }
  }
}

// ---------------- kernel 4: flash attention (causal), one wave = 16 queries
__global__ __launch_bounds__(256) void flash_kernel(
    const _Float16* __restrict__ qh, const _Float16* __restrict__ kh,
    const _Float16* __restrict__ vhT, float* __restrict__ out) {
  __shared__ __align__(32) _Float16 pbuf[8][16 * 40]; // padded stride 40 halfs
  int lane = threadIdx.x & 31;
  int wave = threadIdx.x >> 5;
  int g = lane >> 4, ln = lane & 15;
  int bh = blockIdx.y;            // 0..31
  int b = bh >> 4, h = bh & 15;
  int qb = blockIdx.x * 128 + wave * 16;

  const _Float16* Q = qh  + (((size_t)bh) * S_LEN + qb) * DKH;
  const _Float16* K = kh  + ((size_t)bh) * S_LEN * DKH;
  const _Float16* V = vhT + ((size_t)bh) * DKH * S_LEN;
  _Float16* pb = &pbuf[wave][0];

  // Q fragments for dk 0..31 and 32..63 (kept in registers all loop)
  const _Float16* qrow = Q + ln * DKH;
  v16h qf0 = cat8(*(const v8h*)(qrow + g * 16),
                  *(const v8h*)(qrow + g * 16 + 8));
  v16h qf1 = cat8(*(const v8h*)(qrow + 32 + g * 16),
                  *(const v8h*)(qrow + 40 + g * 16));

  float mrow[8], lrow[8];
  v8f o[4] = {};
#pragma unroll
  for (int r = 0; r < 8; ++r) { mrow[r] = -1e30f; lrow[r] = 0.0f; }

  const int jmax = qb + 15; // last key needed (causal)
  for (int j0 = 0; j0 <= jmax; j0 += 32) {
    const _Float16* k0p = K + (size_t)(j0 + ln) * DKH;
    const _Float16* k1p = K + (size_t)(j0 + 16 + ln) * DKH;
    if (j0 + 32 <= jmax) { // prefetch next key block (global_prefetch_b8)
      __builtin_prefetch(K + (size_t)(j0 + 32 + ln) * DKH, 0, 0);
      __builtin_prefetch(K + (size_t)(j0 + 48 + ln) * DKH, 0, 0);
    }

    // S = Q K^T : 16 queries x 32 keys (two 16-col tiles, 2 K-steps each)
    v8f s0 = {}, s1 = {};
    s0 = wmma16(qf0, *(const v16h*)(k0p + g * 16), s0);
    s0 = wmma16(qf1, *(const v16h*)(k0p + 32 + g * 16), s0);
    s1 = wmma16(qf0, *(const v16h*)(k1p + g * 16), s1);
    s1 = wmma16(qf1, *(const v16h*)(k1p + 32 + g * 16), s1);

    // online softmax; lane holds rows r+8g, col ln (and ln+16)
#pragma unroll
    for (int r = 0; r < 8; ++r) {
      int row = qb + r + 8 * g;
      float v0 = (j0 + ln      <= row) ? s0[r] * 0.125f : -1e30f;
      float v1 = (j0 + 16 + ln <= row) ? s1[r] * 0.125f : -1e30f;
      float mx = rowmax16(fmaxf(v0, v1));
      float mnew = fmaxf(mrow[r], mx);
      float corr = __expf(mrow[r] - mnew);
      float p0 = __expf(v0 - mnew);
      float p1 = __expf(v1 - mnew);
      lrow[r] = lrow[r] * corr + rowsum16(p0 + p1);
      mrow[r] = mnew;
#pragma unroll
      for (int nt = 0; nt < 4; ++nt) o[nt][r] *= corr;
      pb[(r + 8 * g) * 40 + ln]      = (_Float16)p0;
      pb[(r + 8 * g) * 40 + 16 + ln] = (_Float16)p1;
    }

    // P tile (16x32) read back in A-fragment layout from LDS
    v16h pa = cat8(*(const v8h*)(pb + ln * 40 + g * 16),
                   *(const v8h*)(pb + ln * 40 + g * 16 + 8));

    // O += P @ V : B fragment col d = nt*16+ln, K = keys j0 + g*16 + 0..15
#pragma unroll
    for (int nt = 0; nt < 4; ++nt) {
      v16h vf = *(const v16h*)(V + (size_t)(nt * 16 + ln) * S_LEN + j0 + g * 16);
      o[nt] = wmma16(pa, vf, o[nt]);
    }
  }

  // epilogue: out[b, row, h*64 + d] = O / l
#pragma unroll
  for (int r = 0; r < 8; ++r) {
    int row = qb + r + 8 * g;
    float inv = 1.0f / lrow[r];
    size_t base = ((size_t)b * S_LEN + row) * D_DIM + h * DKH;
#pragma unroll
    for (int nt = 0; nt < 4; ++nt)
      out[base + nt * 16 + ln] = o[nt][r] * inv;
  }
}

extern "C" void kernel_launch(void* const* d_in, const int* in_sizes, int n_in,
                              void* d_out, int out_size, void* d_ws, size_t ws_size,
                              hipStream_t stream) {
  const float* q  = (const float*)d_in[0];
  const float* k  = (const float*)d_in[1];
  const float* v  = (const float*)d_in[2];
  // d_in[3] is the causal (tril) mask; causality is applied analytically.
  const float* Wq = (const float*)d_in[4];
  const float* bq = (const float*)d_in[5];
  const float* Wk = (const float*)d_in[6];
  const float* bk = (const float*)d_in[7];
  const float* Wv = (const float*)d_in[8];
  const float* bv = (const float*)d_in[9];

  // workspace layout (bytes): x16[3][4096][1024]f16, w16t[3][1024][1024]f16,
  // qh/kh [B,H,S,DK]f16, vhT [B,H,DK,S]f16  -> ~57 MB total
  char* ws = (char*)d_ws;
  _Float16* x16  = (_Float16*)(ws);
  _Float16* w16t = (_Float16*)(ws + 25165824);
  _Float16* qh   = (_Float16*)(ws + 31457280);
  _Float16* kh   = (_Float16*)(ws + 39845888);
  _Float16* vhT  = (_Float16*)(ws + 48234496);

  cvt_x_kernel<<<dim3(12288), 256, 0, stream>>>(q, k, v, x16);
  cvt_w_kernel<<<dim3(12288), 256, 0, stream>>>(Wq, Wk, Wv, w16t);
  qkv_gemm_kernel<<<dim3(16, 16, 3), 256, 0, stream>>>(x16, w16t, bq, bk, bv,
                                                       qh, kh, vhT);
  flash_kernel<<<dim3(16, 32), 256, 0, stream>>>(qh, kh, vhT, (float*)d_out);
}
